// SelfAttentionV0_34935263986392
// MI455X (gfx1250) — matrix operand order repaired and checked by reading, not compiled, not run on hardware
//
#include <hip/hip_runtime.h>
#include <stdint.h>
#include <stddef.h>

typedef __attribute__((ext_vector_type(16))) __bf16       v16bf;
typedef __attribute__((ext_vector_type(8)))  float        v8f;
typedef __attribute__((ext_vector_type(4)))  unsigned int v4u;
typedef __attribute__((ext_vector_type(4)))  float        v4f;

static constexpr int SEQ = 4096;
static constexpr int DIM = 1024;

#ifndef __has_builtin
#define __has_builtin(x) 0
#endif

#if defined(__AMDGCN__) && __has_builtin(__builtin_amdgcn_tensor_load_to_lds) && \
    __has_builtin(__builtin_amdgcn_s_wait_tensorcnt)
#define USE_TDM 1
#else
#define USE_TDM 0
#endif

// float -> bf16 (round-to-nearest-even), returned as raw u16 for easy packing
__device__ __forceinline__ unsigned short f2bf_u(float f) {
    unsigned int u = __builtin_bit_cast(unsigned int, f);
    u += 0x7FFFu + ((u >> 16) & 1u);
    return (unsigned short)(u >> 16);
}

union FragBF {
    v16bf m;
    v4u   q[2];
};

// ---------------------------------------------------------------------------
// Cast fp32 -> bf16, 8 elements per thread
// ---------------------------------------------------------------------------
__global__ void attn_cast_bf16(const float* __restrict__ in,
                               unsigned short* __restrict__ out, int n8) {
    int i = blockIdx.x * blockDim.x + threadIdx.x;
    if (i >= n8) return;
    const v4f* p = (const v4f*)in + (size_t)i * 2;
    v4f a = p[0];
    v4f b = p[1];
    v4u o;
    o.x = (unsigned)f2bf_u(a.x) | ((unsigned)f2bf_u(a.y) << 16);
    o.y = (unsigned)f2bf_u(a.z) | ((unsigned)f2bf_u(a.w) << 16);
    o.z = (unsigned)f2bf_u(b.x) | ((unsigned)f2bf_u(b.y) << 16);
    o.w = (unsigned)f2bf_u(b.z) | ((unsigned)f2bf_u(b.w) << 16);
    ((v4u*)out)[i] = o;
}

// ---------------------------------------------------------------------------
// Transpose + cast: Wt[n][k] = bf16(W[k][n]).  R = rows of W, C = cols of W.
// ---------------------------------------------------------------------------
__global__ void attn_transpose_cast(const float* __restrict__ W,
                                    unsigned short* __restrict__ Wt,
                                    int R, int C) {
    __shared__ float t[32][33];
    const int bx = blockIdx.x * 32;
    const int by = blockIdx.y * 32;
    const int tx = threadIdx.x;
    const int ty = threadIdx.y;
#pragma unroll
    for (int j = 0; j < 32; j += 8)
        t[ty + j][tx] = W[(size_t)(by + ty + j) * C + (bx + tx)];
    __syncthreads();
#pragma unroll
    for (int j = 0; j < 32; j += 8)
        Wt[(size_t)(bx + ty + j) * R + (by + tx)] = f2bf_u(t[tx][ty + j]);
}

// ---------------------------------------------------------------------------
// WMMA compute for one 128x128x64 K-step from (padded) LDS tiles.
// LDS row pitch LP = 72 ushorts (64 data + 8 pad) -> conflict-free ds_load_b128.
// ---------------------------------------------------------------------------
__device__ __forceinline__ void gemm_compute_64(
    const unsigned short* __restrict__ Asb,
    const unsigned short* __restrict__ Bsb,
    v8f (&acc)[4][2],
    int waveM, int waveN, int lr, int half) {
    constexpr int LP = 72;
#pragma unroll
    for (int kk = 0; kk < 64; kk += 32) {
        // B fragments: lane<16 -> N=lr, K=kk+{0..7,16..23} ; lane>=16 -> +8
        FragBF fb[2];
#pragma unroll
        for (int ni = 0; ni < 2; ++ni) {
            const int col = waveN * 32 + ni * 16 + lr;
            const int kB0 = kk + (half ? 8 : 0);
            fb[ni].q[0] = *(const v4u*)&Bsb[col * LP + kB0];
            fb[ni].q[1] = *(const v4u*)&Bsb[col * LP + kB0 + 16];
        }
        // A fragments: lane<16 -> M=lr, K=kk+{0..7,16..23}; lane>=16 -> +8
#pragma unroll
        for (int mi = 0; mi < 4; ++mi) {
            FragBF fa;
            const int row = waveM * 64 + mi * 16 + lr;
            const int kA0 = kk + (half ? 8 : 0);
            fa.q[0] = *(const v4u*)&Asb[row * LP + kA0];
            fa.q[1] = *(const v4u*)&Asb[row * LP + kA0 + 16];
#pragma unroll
            for (int ni = 0; ni < 2; ++ni) {
                acc[mi][ni] = __builtin_amdgcn_wmma_f32_16x16x32_bf16(
                    false, fa.m, false, fb[ni].m,
                    (short)0, acc[mi][ni], false, false);
            }
        }
    }
}

#if USE_TDM
typedef __attribute__((ext_vector_type(4))) unsigned int u32x4;
typedef __attribute__((ext_vector_type(8))) int          i32x8;
typedef __attribute__((ext_vector_type(4))) int          i32x4;

// Post one TDM 2-D tile load: 128 rows x 64 bf16 elements, row-major source
// with stride tensorD0 elements; hardware pads LDS by 16B after every 128B
// (pad_interval=4 -> 32 DWORDs, pad_amount=3 -> 4 DWORDs) => LDS pitch 144B.
__device__ __forceinline__ void tdm_load_tile128x64(
    const unsigned short* tileStart, unsigned ldsByteOff,
    unsigned tensorD0, unsigned tensorD1) {
    unsigned long long ga = (unsigned long long)(size_t)tileStart;
    u32x4 g0;
    g0[0] = 1u;                                    // count = 1 valid descriptor
    g0[1] = ldsByteOff;                            // lds_addr
    g0[2] = (unsigned)ga;                          // global_addr lo
    g0[3] = (unsigned)(ga >> 32) | (2u << 30);     // global_addr hi | type=2
    i32x8 g1;
    g1[0] = (int)((1u << 16) |                     // data_size = 2 bytes
                  (1u << 20) |                     // pad_enable
                  (4u << 22) |                     // pad_interval: 32 DWORDs
                  (3u << 25));                     // pad_amount: 4 DWORDs
    g1[1] = (int)((tensorD0 & 0xFFFFu) << 16);                                  // tensor_dim0[15:0]
    g1[2] = (int)(((tensorD0 >> 16) & 0xFFFFu) | ((tensorD1 & 0xFFFFu) << 16)); // dim0 hi | dim1 lo
    g1[3] = (int)(((tensorD1 >> 16) & 0xFFFFu) | (64u << 16));                  // dim1 hi | tile_dim0=64
    g1[4] = 128;                                   // tile_dim1 = 128 rows
    g1[5] = (int)tensorD0;                         // tensor_dim0_stride lo (elements)
    g1[6] = 0;                                     // stride0 hi | stride1 lo
    g1[7] = 0;
    i32x4 z4 = {0, 0, 0, 0};
    i32x8 z8 = {0, 0, 0, 0, 0, 0, 0, 0};
    __builtin_amdgcn_tensor_load_to_lds(g0, g1, z4, z4, z8, 0);
}
#endif

// ---------------------------------------------------------------------------
// bf16 WMMA GEMM:  C[M x N] = A[M x K] * Bt[N x K]^T   (both row-major bf16)
// Block tile 128x128x64, 256 threads = 8 waves (2x4), wave tile 64x32.
// TDM path: wave 0 posts tensor_load_to_lds descriptors, double-buffered LDS,
// s_wait_tensorcnt for completion. Fallback: VGPR-staged loads + ds_store.
// ---------------------------------------------------------------------------
template <bool OUT_BF16, bool TRANS_OUT>
__global__ __launch_bounds__(256)
void attn_gemm_bf16(const unsigned short* __restrict__ A,
                    const unsigned short* __restrict__ Bt,
                    float* __restrict__ Cf,
                    unsigned short* __restrict__ Cb,
                    int M, int N, int Kd, float outScale) {
    constexpr int BM = 128, BN = 128, BK = 64;
    constexpr int LP = BK + 8;  // padded LDS row pitch (ushorts)

    const int tid   = threadIdx.x;
    const int wave  = tid >> 5;
    const int lane  = tid & 31;
    const int waveM = wave >> 2;  // 0..1
    const int waveN = wave & 3;   // 0..3
    const int lr    = lane & 15;
    const int half  = lane >> 4;

    const int m0 = blockIdx.y * BM;
    const int n0 = blockIdx.x * BN;

    v8f zero = {0.f, 0.f, 0.f, 0.f, 0.f, 0.f, 0.f, 0.f};
    v8f acc[4][2];
#pragma unroll
    for (int mi = 0; mi < 4; ++mi)
#pragma unroll
        for (int ni = 0; ni < 2; ++ni) acc[mi][ni] = zero;

#if USE_TDM
    __shared__ __align__(16) unsigned short As[2][BM * LP];
    __shared__ __align__(16) unsigned short Bs[2][BN * LP];

    const int nk = Kd / BK;
    if (wave == 0) {
        tdm_load_tile128x64(&A[(size_t)m0 * Kd], (unsigned)(size_t)&As[0][0],
                            (unsigned)Kd, (unsigned)M);
        tdm_load_tile128x64(&Bt[(size_t)n0 * Kd], (unsigned)(size_t)&Bs[0][0],
                            (unsigned)Kd, (unsigned)N);
    }
    for (int i = 0; i < nk; ++i) {
        const int cur = i & 1;
        if (wave == 0) {
            if (i + 1 < nk) {
                const int nxt = cur ^ 1;
                tdm_load_tile128x64(&A[(size_t)m0 * Kd + (size_t)(i + 1) * BK],
                                    (unsigned)(size_t)&As[nxt][0],
                                    (unsigned)Kd, (unsigned)M);
                tdm_load_tile128x64(&Bt[(size_t)n0 * Kd + (size_t)(i + 1) * BK],
                                    (unsigned)(size_t)&Bs[nxt][0],
                                    (unsigned)Kd, (unsigned)N);
                __builtin_amdgcn_s_wait_tensorcnt(2);  // current stage landed
            } else {
                __builtin_amdgcn_s_wait_tensorcnt(0);  // last stage landed
            }
        }
        __syncthreads();
        gemm_compute_64(&As[cur][0], &Bs[cur][0], acc, waveM, waveN, lr, half);
        __syncthreads();
    }
#else
    __shared__ __align__(16) unsigned short As[BM * LP];
    __shared__ __align__(16) unsigned short Bs[BN * LP];

    const int ldRow = tid >> 3;        // 0..31
    const int ldCol = (tid & 7) * 8;   // 0..56, 8 ushorts = 16 B

    for (int kb = 0; kb < Kd; kb += BK) {
#pragma unroll
        for (int p = 0; p < 4; ++p) {
            const int r = ldRow + 32 * p;
            *(v4u*)&As[r * LP + ldCol] =
                *(const v4u*)&A[(size_t)(m0 + r) * Kd + kb + ldCol];
            *(v4u*)&Bs[r * LP + ldCol] =
                *(const v4u*)&Bt[(size_t)(n0 + r) * Kd + kb + ldCol];
        }
        __syncthreads();
        gemm_compute_64(As, Bs, acc, waveM, waveN, lr, half);
        __syncthreads();
    }
#endif

    // C/D layout: VGPR r, lane<16 -> (M=r, N=lane); lane>=16 -> (M=8+r, N=lane-16)
    const int hb = half * 8;
#pragma unroll
    for (int mi = 0; mi < 4; ++mi) {
#pragma unroll
        for (int ni = 0; ni < 2; ++ni) {
            const int mb = m0 + waveM * 64 + mi * 16 + hb;
            const int nn = n0 + waveN * 32 + ni * 16 + lr;
#pragma unroll
            for (int r = 0; r < 8; ++r) {
                const float v = acc[mi][ni][r] * outScale;
                if (OUT_BF16) {
                    if (TRANS_OUT)
                        Cb[(size_t)nn * M + (mb + r)] = f2bf_u(v);
                    else
                        Cb[(size_t)(mb + r) * N + nn] = f2bf_u(v);
                } else {
                    Cf[(size_t)(mb + r) * N + nn] = v;
                }
            }
        }
    }
}

// ---------------------------------------------------------------------------
// Row softmax: one 256-thread workgroup per row of S (N = 4096), emit bf16 P.
// ---------------------------------------------------------------------------
__global__ __launch_bounds__(256)
void attn_softmax_rows(const float* __restrict__ S,
                       unsigned short* __restrict__ P, int N) {
    __shared__ float red[256];
    const int row = blockIdx.x;
    const int tid = threadIdx.x;
    const float* s = S + (size_t)row * N;

    float v[16];
#pragma unroll
    for (int i = 0; i < 16; ++i) v[i] = s[i * 256 + tid];

    float m = v[0];
#pragma unroll
    for (int i = 1; i < 16; ++i) m = fmaxf(m, v[i]);
    red[tid] = m;
    __syncthreads();
    for (int off = 128; off > 0; off >>= 1) {
        if (tid < off) red[tid] = fmaxf(red[tid], red[tid + off]);
        __syncthreads();
    }
    const float rowmax = red[0];
    __syncthreads();

    float e[16];
    float sum = 0.f;
#pragma unroll
    for (int i = 0; i < 16; ++i) {
        e[i] = __expf(v[i] - rowmax);
        sum += e[i];
    }
    red[tid] = sum;
    __syncthreads();
    for (int off = 128; off > 0; off >>= 1) {
        if (tid < off) red[tid] += red[tid + off];
        __syncthreads();
    }
    const float inv = 1.0f / red[0];

    unsigned short* p = P + (size_t)row * N;
#pragma unroll
    for (int i = 0; i < 16; ++i) p[i * 256 + tid] = f2bf_u(e[i] * inv);
}

// ---------------------------------------------------------------------------
extern "C" void kernel_launch(void* const* d_in, const int* in_sizes, int n_in,
                              void* d_out, int out_size, void* d_ws, size_t ws_size,
                              hipStream_t stream) {
    const float* x  = (const float*)d_in[0];
    const float* Wq = (const float*)d_in[1];
    const float* Wk = (const float*)d_in[2];
    const float* Wv = (const float*)d_in[3];
    float* out = (float*)d_out;

    char* ws = (char*)d_ws;
    size_t off = 0;
    auto alloc = [&](size_t bytes) -> char* {
        char* p = ws + off;
        off += (bytes + 255) & ~(size_t)255;
        return p;
    };
    unsigned short* Xb  = (unsigned short*)alloc((size_t)SEQ * DIM * 2);
    unsigned short* Wqt = (unsigned short*)alloc((size_t)DIM * DIM * 2);
    unsigned short* Wkt = (unsigned short*)alloc((size_t)DIM * DIM * 2);
    unsigned short* Wvt = (unsigned short*)alloc((size_t)DIM * DIM * 2);
    unsigned short* Qb  = (unsigned short*)alloc((size_t)SEQ * DIM * 2);
    unsigned short* Kb  = (unsigned short*)alloc((size_t)SEQ * DIM * 2);
    unsigned short* Vt  = (unsigned short*)alloc((size_t)DIM * SEQ * 2);  // [DIM x SEQ]
    float*          S   = (float*)         alloc((size_t)SEQ * SEQ * 4);
    unsigned short* Pb  = (unsigned short*)alloc((size_t)SEQ * SEQ * 2);

    // 1) casts / transposes
    {
        const int n8 = SEQ * DIM / 8;
        attn_cast_bf16<<<(n8 + 255) / 256, 256, 0, stream>>>(x, Xb, n8);
        dim3 tb(32, 8), tg(DIM / 32, DIM / 32);
        attn_transpose_cast<<<tg, tb, 0, stream>>>(Wq, Wqt, DIM, DIM);
        attn_transpose_cast<<<tg, tb, 0, stream>>>(Wk, Wkt, DIM, DIM);
        attn_transpose_cast<<<tg, tb, 0, stream>>>(Wv, Wvt, DIM, DIM);
    }

    // 2) QKV projections (Q carries 1/sqrt(d)=1/32; V stored transposed)
    {
        dim3 g(DIM / 128, SEQ / 128);
        attn_gemm_bf16<true, false><<<g, 256, 0, stream>>>(Xb, Wqt, nullptr, Qb, SEQ, DIM, DIM, 0.03125f);
        attn_gemm_bf16<true, false><<<g, 256, 0, stream>>>(Xb, Wkt, nullptr, Kb, SEQ, DIM, DIM, 1.0f);
        attn_gemm_bf16<true, true ><<<g, 256, 0, stream>>>(Xb, Wvt, nullptr, Vt, SEQ, DIM, DIM, 1.0f);
    }

    // 3) scores S = (Q/32) @ K^T (f32 accumulate + out)
    {
        dim3 g(SEQ / 128, SEQ / 128);
        attn_gemm_bf16<false, false><<<g, 256, 0, stream>>>(Qb, Kb, S, nullptr, SEQ, SEQ, DIM, 1.0f);
    }

    // 4) softmax rows -> bf16 probabilities
    attn_softmax_rows<<<SEQ, 256, 0, stream>>>(S, Pb, SEQ);

    // 5) out = P @ V  (A = P [SEQ x SEQ], Bt = V^T [DIM x SEQ])
    {
        dim3 g(DIM / 128, SEQ / 128);
        attn_gemm_bf16<false, false><<<g, 256, 0, stream>>>(Pb, Vt, out, nullptr, SEQ, DIM, SEQ, 1.0f);
    }
}
